// GradientDifferenceLossCanny_v1_84739704750348
// MI455X (gfx1250) — compile-verified, matrix operand order checked by reading, not run on hardware
//
#include <hip/hip_runtime.h>
#include <stdint.h>

#define H 512
#define W 512
#define WPR 16                  // 512 bits / 32 = 16 words per row
#define IMG_WORDS (H * WPR)     // 8192 words = 32 KB per bit image
#define NIMG 96                 // 32 batch * 3 channels per input
#define TOTAL_IMG 192           // pred + target
#define LOW_T 50.0f
#define HIGH_T 150.0f

typedef __attribute__((ext_vector_type(8))) int v8i;

// LDS byte offset of a generic pointer to __shared__ (LDS aperture: addr[31:0] = LDS offset)
__device__ __forceinline__ uint32_t lds_off(const void* p) {
    return (uint32_t)(uintptr_t)p;
}

// async global -> LDS, per-lane, 4 bytes (ASYNCcnt)
__device__ __forceinline__ void async_load_b32(uint32_t ldsoff, const void* gaddr) {
    asm volatile("global_load_async_to_lds_b32 %0, %1, off"
                 :: "v"(ldsoff), "v"(gaddr) : "memory");
}
// async global -> LDS, per-lane, 16 bytes
__device__ __forceinline__ void async_load_b128(uint32_t ldsoff, const void* gaddr) {
    asm volatile("global_load_async_to_lds_b128 %0, %1, off"
                 :: "v"(ldsoff), "v"(gaddr) : "memory");
}
// async LDS -> global, per-lane, 16 bytes
__device__ __forceinline__ void async_store_b128(void* gaddr, uint32_t ldsoff) {
    asm volatile("global_store_async_from_lds_b128 %0, %1, off"
                 :: "v"(gaddr), "v"(ldsoff) : "memory");
}
__device__ __forceinline__ void wait_async0() {
    asm volatile("s_wait_asynccnt 0" ::: "memory");
}

// ---------------------------------------------------------------- K0: zero the mismatch counter
__global__ void k_zero(unsigned* cnt) {
    if (threadIdx.x == 0) *cnt = 0u;
}

// ---------------------------------------------------------------- K1: Sobel + NMS + thresholds -> packed weak/strong bitmaps
// block (32,8) covers a 32x8 pixel tile; one wave32 = one row of 32 pixels -> ballot packs one word.
// Input tile staged via async global->LDS (no VGPR round-trip on the bandwidth-bound stage).
__global__ __launch_bounds__(256) void k_sobel_nms(const float* __restrict__ pred,
                                                   const float* __restrict__ targ,
                                                   uint32_t* __restrict__ wsWeak,
                                                   uint32_t* __restrict__ wsCur) {
    __shared__ float sIn[12][37];   // input tile (y0-2..y0+9) x (x0-2..x0+33), edge-clamped
    __shared__ float sMag[10][35];  // |gx|+|gy| tile (y0-1..y0+8) x (x0-1..x0+32), 0 outside image

    const int iz = blockIdx.z;
    const float* src = (iz < NIMG) ? pred : targ;
    const int img = (iz < NIMG) ? iz : iz - NIMG;
    const size_t base = (size_t)img * (H * W);
    const int x0 = blockIdx.x * 32;
    const int y0 = blockIdx.y * 8;
    const int tid = threadIdx.x + threadIdx.y * 32;

    // hint the next tile-row into cache (global_prefetch_b8)
    __builtin_prefetch(src + base + (size_t)min(y0 + 10, H - 1) * W + x0, 0, 3);

    // stage 12x36 input tile with replicate ('edge') clamp, via async-to-LDS
    for (int i = tid; i < 12 * 36; i += 256) {
        int ly = i / 36, lx = i % 36;
        int gy = min(max(y0 - 2 + ly, 0), H - 1);
        int gx = min(max(x0 - 2 + lx, 0), W - 1);
        async_load_b32(lds_off(&sIn[ly][lx]), src + base + (size_t)gy * W + gx);
    }
    wait_async0();
    __syncthreads();

    // gradient magnitude on 10x34 halo tile (zero-padded outside the image, matching magp)
    for (int i = tid; i < 10 * 34; i += 256) {
        int my = i / 34, mx = i % 34;
        int r = y0 - 1 + my, c = x0 - 1 + mx;
        float m = 0.0f;
        if (r >= 0 && r < H && c >= 0 && c < W) {
            float a00 = sIn[my][mx],     a01 = sIn[my][mx + 1],     a02 = sIn[my][mx + 2];
            float a10 = sIn[my + 1][mx],                            a12 = sIn[my + 1][mx + 2];
            float a20 = sIn[my + 2][mx], a21 = sIn[my + 2][mx + 1], a22 = sIn[my + 2][mx + 2];
            float gx = (a02 - a00) + 2.0f * (a12 - a10) + (a22 - a20);
            float gy = (a20 - a00) + 2.0f * (a21 - a01) + (a22 - a02);
            m = fabsf(gx) + fabsf(gy);
        }
        sMag[my][mx] = m;
    }
    __syncthreads();

    // per-pixel sector + NMS + thresholds
    const int tx = threadIdx.x, ty = threadIdx.y;
    float a00 = sIn[ty + 1][tx + 1], a01 = sIn[ty + 1][tx + 2], a02 = sIn[ty + 1][tx + 3];
    float a10 = sIn[ty + 2][tx + 1],                            a12 = sIn[ty + 2][tx + 3];
    float a20 = sIn[ty + 3][tx + 1], a21 = sIn[ty + 3][tx + 2], a22 = sIn[ty + 3][tx + 3];
    float gx = (a02 - a00) + 2.0f * (a12 - a10) + (a22 - a20);
    float gy = (a20 - a00) + 2.0f * (a21 - a01) + (a22 - a02);
    float mc = sMag[ty + 1][tx + 1];

    const float T1 = 0.41421356237309503f;  // tan(22.5 deg)
    const float T2 = 2.41421356237309500f;  // tan(67.5 deg)
    float ax = fabsf(gx), ay = fabsf(gy);
    float n1, n2;
    if (ay < T1 * ax) {                 // d0: horizontal gradient -> left/right
        n1 = sMag[ty + 1][tx + 2]; n2 = sMag[ty + 1][tx];
    } else if (ay >= T2 * ax) {         // d2: vertical gradient -> up/down
        n1 = sMag[ty][tx + 1];     n2 = sMag[ty + 2][tx + 1];
    } else if (gx * gy > 0.0f) {        // d1: up-right / down-left
        n1 = sMag[ty][tx + 2];     n2 = sMag[ty + 2][tx];
    } else {                            // d3: up-left / down-right
        n1 = sMag[ty][tx];         n2 = sMag[ty + 2][tx + 2];
    }

    bool weak   = (mc >= n1) && (mc >= n2) && (mc > LOW_T);
    bool strong = weak && (mc > HIGH_T);

    unsigned wb = __builtin_amdgcn_ballot_w32(weak);
    unsigned sb = __builtin_amdgcn_ballot_w32(strong);
    if (tx == 0) {
        size_t bidx = (size_t)iz * IMG_WORDS + (size_t)(y0 + ty) * WPR + blockIdx.x;
        wsWeak[bidx] = wb;
        wsCur[bidx]  = sb;   // hysteresis starts from strong
    }
}

// ---------------------------------------------------------------- K2: full-image LDS-resident hysteresis to fixpoint
// One 1024-thread workgroup per image; cur+weak bitmaps (64 KB) live in LDS (320 KB/WGP on CDNA5).
// Bitmaps streamed in/out with async b128 LDS transfers.
__global__ __launch_bounds__(1024) void k_hysteresis(const uint32_t* __restrict__ wsWeak,
                                                     uint32_t* __restrict__ wsCur) {
    extern __shared__ uint32_t lds[];
    uint32_t* cur = lds;
    uint32_t* wk  = lds + IMG_WORDS;
    __shared__ int changed;

    const int t = threadIdx.x;
    const size_t base = (size_t)blockIdx.x * IMG_WORDS;

    // async-stage both 32KB bitmaps: 1024 lanes x 2 chunks x 16B each per buffer
#pragma unroll
    for (int c = 0; c < 2; ++c) {
        int e = (c * 1024 + t) * 4;   // word index, 16B granularity
        async_load_b128(lds_off(cur + e), wsCur + base + e);
        async_load_b128(lds_off(wk + e),  wsWeak + base + e);
    }
    if (t == 0) changed = 0;
    wait_async0();
    __syncthreads();

    const int r = t >> 1;               // row 0..511
    const int halfc = (t & 1) * 8;      // word columns 0..7 or 8..15

    for (int it = 0; it < 4096; ++it) {
        int my = 0;
        for (int w = 0; w < 8; ++w) {
            int col = halfc + w;
            int idx = r * WPR + col;
            uint32_t mid = cur[idx];
            uint32_t wkw = wk[idx];
            if ((wkw & ~mid) == 0u) continue;   // word saturated: nothing left to add
            uint32_t v = mid;
            if (r > 0)     v |= cur[idx - WPR];
            if (r < H - 1) v |= cur[idx + WPR];
            uint32_t vl = 0u, vr = 0u;
            if (col > 0) {
                vl = cur[idx - 1];
                if (r > 0)     vl |= cur[idx - 1 - WPR];
                if (r < H - 1) vl |= cur[idx - 1 + WPR];
            }
            if (col < WPR - 1) {
                vr = cur[idx + 1];
                if (r > 0)     vr |= cur[idx + 1 - WPR];
                if (r < H - 1) vr |= cur[idx + 1 + WPR];
            }
            // 3x3 (8-connected) dilation: vertical OR, then horizontal smear with word carries
            uint32_t dil = v | (v << 1) | (v >> 1) | (vl >> 31) | (vr << 31);
            uint32_t nw = mid | (wkw & dil);
            if (nw != mid) { cur[idx] = nw; my = 1; }   // monotone in-place update is safe
        }
        if (my) atomicOr(&changed, 1);
        __syncthreads();
        int done = (changed == 0);
        __syncthreads();                 // everyone sampled `changed` before reset
        if (done) break;                 // fixpoint certified: a full pass made no writes
        if (t == 0) changed = 0;
        __syncthreads();
    }

    // async-store the converged bitmap back (S_ENDPGM wait-idle also drains ASYNCcnt)
#pragma unroll
    for (int c = 0; c < 2; ++c) {
        int e = (c * 1024 + t) * 4;
        async_store_b128(wsCur + base + e, lds_off(cur + e));
    }
    wait_async0();
}

// ---------------------------------------------------------------- K3: mismatch count via v_wmma_i32_16x16x64_iu8
// lane packs 32 popcounts (each <=32, fits u8) into the 16x64 u8 A operand; B = all-ones
// => sum of all 256 D elements == 16 * sum of packed bytes, independent of the A layout.
__global__ __launch_bounds__(256) void k_reduce_wmma(const uint32_t* __restrict__ wsCur,
                                                     unsigned* __restrict__ cnt) {
    __shared__ unsigned bsum;
    if (threadIdx.x == 0) bsum = 0u;
    __syncthreads();

    const uint32_t* p = wsCur;                                   // pred edge bitmaps
    const uint32_t* q = wsCur + (size_t)NIMG * IMG_WORDS;        // target edge bitmaps
    const int gtid = blockIdx.x * 256 + threadIdx.x;
    const int wave = gtid >> 5;
    const int lane = gtid & 31;
    const size_t wbase = (size_t)wave * 1024 + (size_t)lane * 32; // 768 waves * 1024 words = 786432 total

    v8i a;
#pragma unroll
    for (int v = 0; v < 8; ++v) {
        uint32_t x0 = p[wbase + v * 4 + 0] ^ q[wbase + v * 4 + 0];
        uint32_t x1 = p[wbase + v * 4 + 1] ^ q[wbase + v * 4 + 1];
        uint32_t x2 = p[wbase + v * 4 + 2] ^ q[wbase + v * 4 + 2];
        uint32_t x3 = p[wbase + v * 4 + 3] ^ q[wbase + v * 4 + 3];
        unsigned pk = (unsigned)__popc(x0) | ((unsigned)__popc(x1) << 8) |
                      ((unsigned)__popc(x2) << 16) | ((unsigned)__popc(x3) << 24);
        a[v] = (int)pk;
    }
    v8i b;
#pragma unroll
    for (int v = 0; v < 8; ++v) b[v] = 0x01010101;   // u8 ones
    v8i c = {0, 0, 0, 0, 0, 0, 0, 0};
    // 7 args: (sgn_a, A, sgn_b, B, C, reuse_a, reuse_b); unsigned x unsigned
    c = __builtin_amdgcn_wmma_i32_16x16x64_iu8(false, a, false, b, c, false, false);

    unsigned part = (unsigned)(c[0] + c[1] + c[2] + c[3] + c[4] + c[5] + c[6] + c[7]);
    atomicAdd(&bsum, part);
    __syncthreads();
    if (threadIdx.x == 0) atomicAdd(cnt, bsum);      // accumulates 16 * mismatch_bits
}

// ---------------------------------------------------------------- K4: finalize scalar loss
__global__ void k_finalize(const unsigned* __restrict__ cnt, float* __restrict__ out) {
    if (threadIdx.x == 0 && blockIdx.x == 0) {
        double mism = (double)(*cnt) / 16.0;                 // exact integer
        out[0] = (float)(65025.0 * mism / 25165824.0);       // 255^2 * count / (32*3*512*512)
    }
}

// ---------------------------------------------------------------- launch
extern "C" void kernel_launch(void* const* d_in, const int* in_sizes, int n_in,
                              void* d_out, int out_size, void* d_ws, size_t ws_size,
                              hipStream_t stream) {
    (void)in_sizes; (void)n_in; (void)out_size; (void)ws_size;
    const float* pred = (const float*)d_in[0];
    const float* targ = (const float*)d_in[1];

    uint32_t* wsWeak = (uint32_t*)d_ws;                                  // 192 * 32 KB = 6 MB
    uint32_t* wsCur  = wsWeak + (size_t)TOTAL_IMG * IMG_WORDS;           // 6 MB (strong -> final)
    unsigned* cnt    = (unsigned*)(wsCur + (size_t)TOTAL_IMG * IMG_WORDS);

    k_zero<<<1, 1, 0, stream>>>(cnt);

    dim3 g1(W / 32, H / 8, TOTAL_IMG), b1(32, 8);
    k_sobel_nms<<<g1, b1, 0, stream>>>(pred, targ, wsWeak, wsCur);

    k_hysteresis<<<TOTAL_IMG, 1024, 2 * IMG_WORDS * sizeof(uint32_t), stream>>>(wsWeak, wsCur);

    k_reduce_wmma<<<NIMG, 256, 0, stream>>>(wsCur, cnt);

    k_finalize<<<1, 1, 0, stream>>>(cnt, (float*)d_out);
}
